// kv_cache_group_query_49383533970034
// MI455X (gfx1250) — matrix operand order repaired and checked by reading, not compiled, not run on hardware
//
#include <hip/hip_runtime.h>
#include <hip/hip_bf16.h>
#include <cstdint>

// ---------------------------------------------------------------------------
// GQA attention block for MI455X (gfx1250, wave32, WMMA).
// All GEMMs use v_wmma_f32_16x16x32_bf16 (bf16 in, f32 accumulate).
// ---------------------------------------------------------------------------

#define EMBED_DIM   4096
#define NUM_Q_HEADS 32
#define NUM_KV_HEADS 8
#define HEAD_DIM    128
#define GROUP       4
#define KV_CAP      2048      // 2 * KV_SEQ_LEN
#define BATCH       2
#define SEQ         1024

typedef __attribute__((ext_vector_type(16))) __bf16 v16bf;
typedef __attribute__((ext_vector_type(8)))  __bf16 v8bf;
typedef __attribute__((ext_vector_type(8)))  float  v8f;
typedef int v4i __attribute__((vector_size(16)));
typedef __attribute__((address_space(1))) v4i* gptr_v4i;
typedef __attribute__((address_space(3))) v4i* lptr_v4i;

// Async global->LDS staging path (ASYNCcnt) when the toolchain exposes it.
#if defined(__has_builtin)
# if __has_builtin(__builtin_amdgcn_global_load_async_to_lds_b128)
#  define HAVE_ASYNC_LDS 1
# endif
# if __has_builtin(__builtin_amdgcn_s_wait_asynccnt)
#  define HAVE_WAIT_ASYNC_BUILTIN 1
# endif
#endif
#ifndef HAVE_ASYNC_LDS
# define HAVE_ASYNC_LDS 0
#endif
#ifndef HAVE_WAIT_ASYNC_BUILTIN
# define HAVE_WAIT_ASYNC_BUILTIN 0
#endif

static __device__ __forceinline__ void wait_asynccnt0() {
#if HAVE_WAIT_ASYNC_BUILTIN
  __builtin_amdgcn_s_wait_asynccnt(0);
#else
  asm volatile("s_wait_asynccnt 0x0" ::: "memory");
#endif
}

// ---------------------------------------------------------------------------
// f32 -> bf16 conversion (weights, activations, cache)
// ---------------------------------------------------------------------------
__global__ void cvt_f32_to_bf16(const float* __restrict__ in,
                                __bf16* __restrict__ out, int n) {
  int i = blockIdx.x * blockDim.x + threadIdx.x;
  if (i < n) out[i] = (__bf16)in[i];
}

// A-fragment loader (16x32 bf16, contiguous k order): lane r holds row m; half 0 ->
// K {0..15}, half 1 -> {16..31}. Two 16-byte vector loads.
static __device__ __forceinline__ v16bf load_a_frag(const __bf16* p) {
  v8bf lo = *(const v8bf*)(p);
  v8bf hi = *(const v8bf*)(p + 8);
  v16bf a;
#pragma unroll
  for (int i = 0; i < 8; ++i) { a[i] = lo[i]; a[i + 8] = hi[i]; }
  return a;
}

// ---------------------------------------------------------------------------
// GEMM: C[M,N] (f32) = A[M,K] (bf16, row-major) * W[N,K]^T (bf16, row-major)
// Block = 256 threads = 8 waves. Block tile: 64 M x 256 N.
// Wave tile: 32 M x 64 N -> 2 A-frags, 4 B-frags, 8 WMMAs per k-step.
// All 12 b128 loads for a k-step are issued before the WMMA chain so the
// loads overlap the 8-deep matrix-op chain (B frags reused by both M rows).
// ---------------------------------------------------------------------------
__global__ __launch_bounds__(256) void gemm_nt_bf16(
    const __bf16* __restrict__ A, const __bf16* __restrict__ W,
    float* __restrict__ C, int M, int N, int K) {
  const int lane = threadIdx.x & 31;
  const int wave = threadIdx.x >> 5;
  const int r  = lane & 15;
  const int hf = lane >> 4;

  const int m0 = blockIdx.x * 64 + (wave & 1) * 32;
  const int n0 = blockIdx.y * 256 + (wave >> 1) * 64;

  v8f acc[2][4];
#pragma unroll
  for (int mt = 0; mt < 2; ++mt)
#pragma unroll
    for (int nt = 0; nt < 4; ++nt)
#pragma unroll
      for (int i = 0; i < 8; ++i) acc[mt][nt][i] = 0.0f;

  const __bf16* arow0 = A + (size_t)(m0 + r) * K;
  const __bf16* arow1 = A + (size_t)(m0 + 16 + r) * K;
  const __bf16* wrow  = W + (size_t)(n0 + r) * K;

  for (int k0 = 0; k0 < K; k0 += 32) {
    __builtin_prefetch(arow0 + k0 + 256, 0, 3);   // global_prefetch_b8
    __builtin_prefetch(wrow + k0 + 256, 0, 3);

    v16bf a0 = load_a_frag(arow0 + k0 + hf * 16);
    v16bf a1 = load_a_frag(arow1 + k0 + hf * 16);
    v16bf b[4];
#pragma unroll
    for (int nt = 0; nt < 4; ++nt)
      b[nt] = *(const v16bf*)(wrow + (size_t)nt * 16 * K + k0 + hf * 16);

#pragma unroll
    for (int nt = 0; nt < 4; ++nt)
      acc[0][nt] = __builtin_amdgcn_wmma_f32_16x16x32_bf16(
          false, a0, false, b[nt], (short)0, acc[0][nt], false, false);
#pragma unroll
    for (int nt = 0; nt < 4; ++nt)
      acc[1][nt] = __builtin_amdgcn_wmma_f32_16x16x32_bf16(
          false, a1, false, b[nt], (short)0, acc[1][nt], false, false);
  }

  // C/D layout: lanes 0-15 -> M=j, N=lane; lanes 16-31 -> M=j+8, N=lane-16.
#pragma unroll
  for (int mt = 0; mt < 2; ++mt)
#pragma unroll
    for (int nt = 0; nt < 4; ++nt)
#pragma unroll
      for (int i = 0; i < 8; ++i)
        C[(size_t)(m0 + mt * 16 + i + hf * 8) * N + (n0 + nt * 16 + r)] =
            acc[mt][nt][i];
}

// ---------------------------------------------------------------------------
// RoPE on Q (f32 in, bf16 out, same [b,t,h,d] layout).
// ---------------------------------------------------------------------------
__global__ void rope_q_kernel(const float* __restrict__ qf,
                              const int* __restrict__ spp,
                              __bf16* __restrict__ qh) {
  int gid = blockIdx.x * blockDim.x + threadIdx.x;  // BATCH*SEQ*32*64
  int i = gid & 63;
  int h = (gid >> 6) & 31;
  int t = (gid >> 11) & 1023;
  int b = gid >> 21;
  int pos = *spp + t;
  float inv = __expf(-(float)i * (9.210340371976184f / 64.0f)); // theta^-(i/64)
  float s, c;
  __sincosf((float)pos * inv, &s, &c);
  size_t base = ((size_t)(b * SEQ + t)) * EMBED_DIM + h * HEAD_DIM + 2 * i;
  float x0 = qf[base], x1 = qf[base + 1];
  qh[base]     = (__bf16)(x0 * c - x1 * s);
  qh[base + 1] = (__bf16)(x0 * s + x1 * c);
}

// RoPE on K, writing into bf16 cache layout [B, 2048, 8, 128] at start_pos+t.
__global__ void rope_k_kernel(const float* __restrict__ kf,
                              const int* __restrict__ spp,
                              __bf16* __restrict__ kh) {
  int gid = blockIdx.x * blockDim.x + threadIdx.x;  // BATCH*SEQ*8*64
  int i = gid & 63;
  int h = (gid >> 6) & 7;
  int t = (gid >> 9) & 1023;
  int b = gid >> 19;
  int sp = *spp;
  int pos = sp + t;
  float inv = __expf(-(float)i * (9.210340371976184f / 64.0f));
  float s, c;
  __sincosf((float)pos * inv, &s, &c);
  size_t src = ((size_t)(b * SEQ + t)) * (NUM_KV_HEADS * HEAD_DIM) + h * HEAD_DIM + 2 * i;
  size_t dst = (((size_t)(b * KV_CAP + sp + t)) * NUM_KV_HEADS + h) * HEAD_DIM + 2 * i;
  float x0 = kf[src], x1 = kf[src + 1];
  kh[dst]     = (__bf16)(x0 * c - x1 * s);
  kh[dst + 1] = (__bf16)(x0 * s + x1 * c);
}

// V: convert and scatter into cache layout at start_pos+t.
__global__ void v_store_kernel(const float* __restrict__ vf,
                               const int* __restrict__ spp,
                               __bf16* __restrict__ vh) {
  int gid = blockIdx.x * blockDim.x + threadIdx.x;  // BATCH*SEQ*8*128
  int d = gid & 127;
  int h = (gid >> 7) & 7;
  int t = (gid >> 10) & 1023;
  int b = gid >> 20;
  int sp = *spp;
  size_t src = ((size_t)(b * SEQ + t)) * (NUM_KV_HEADS * HEAD_DIM) + h * HEAD_DIM + d;
  size_t dst = (((size_t)(b * KV_CAP + sp + t)) * NUM_KV_HEADS + h) * HEAD_DIM + d;
  vh[dst] = (__bf16)vf[src];
}

// ---------------------------------------------------------------------------
// Attention: one block (4 waves, 128 threads) per (b, q-head, 16-row q tile).
// Dynamic LDS (~137 KB; CDNA5 WGP has 320 KB):
//   sc[16][2048] f32 score tile | red8[16][8] | red1[16] | vstage[32][128] bf16
//  Phase 1: S = Q K^T  (WMMA, causal mask + scale in register C-layout)
//  Phase 2: row max / exp / sum via float4 LDS scans (normalization deferred)
//  Phase 3: ctx = P V  with V chunks staged to LDS via async-to-LDS loads
//           (ASYNCcnt path) and probs converted f32->bf16 on the fly;
//           1/sum folded into the final store.
// ---------------------------------------------------------------------------
__global__ __launch_bounds__(128) void attn_kernel(
    const __bf16* __restrict__ qh, const __bf16* __restrict__ kh,
    const __bf16* __restrict__ vh, const int* __restrict__ spp,
    __bf16* __restrict__ ctxh) {
  extern __shared__ float sc[];                    // [16][2048]
  float*  red8   = sc + 16 * KV_CAP;               // [16][8]
  float*  red1   = red8 + 16 * 8;                  // [16]
  __bf16* vstage = (__bf16*)(red1 + 16);           // [32][128], 16B aligned

  const int tid  = threadIdx.x;
  const int lane = tid & 31;
  const int wave = tid >> 5;
  const int r    = lane & 15;
  const int hf   = lane >> 4;

  const int qt = blockIdx.x & 63;
  const int h  = (blockIdx.x >> 6) & 31;
  const int b  = blockIdx.x >> 11;
  const int kvh = h >> 2;                          // GROUP = 4
  const int sp = *spp;
  const int S  = sp + SEQ;                         // end_pos
  const int qbase = qt * 16;
  const float scale = 0.08838834764831845f;        // 1/sqrt(128)

  // Q fragments: 4 chunks of K=32 covering HEAD_DIM=128. A-matrix layout.
  v16bf qa[4];
  {
    const __bf16* qrow =
        qh + ((size_t)(b * SEQ + qbase + r)) * EMBED_DIM + h * HEAD_DIM;
#pragma unroll
    for (int kk = 0; kk < 4; ++kk)
      qa[kk] = load_a_frag(qrow + kk * 32 + hf * 16);
  }

  // --- Phase 1: scores, padded to a multiple of 32 keys -------------------
  const int nkt = ((S + 31) >> 5) << 1;    // 16-key tiles, rounded to pairs
  const int Spad = nkt * 16;               // <= 2048
  for (int kt = wave; kt < nkt; kt += 4) { // wave-uniform: EXEC stays full
    int s0 = kt * 16;
    v8f acc;
#pragma unroll
    for (int i = 0; i < 8; ++i) acc[i] = 0.0f;
    const __bf16* krow =
        kh + (((size_t)(b * KV_CAP + s0 + r)) * NUM_KV_HEADS + kvh) * HEAD_DIM;
    v16bf kb[4];                           // preload, then chain 4 WMMAs
#pragma unroll
    for (int kk = 0; kk < 4; ++kk)
      kb[kk] = *(const v16bf*)(krow + kk * 32 + hf * 16);
#pragma unroll
    for (int kk = 0; kk < 4; ++kk)
      acc = __builtin_amdgcn_wmma_f32_16x16x32_bf16(
          false, qa[kk], false, kb[kk], (short)0, acc, false, false);
#pragma unroll
    for (int i = 0; i < 8; ++i) {
      int m = i + hf * 8;                  // local query row
      int s = s0 + r;                      // key index
      float v = acc[i] * scale;
      v = (s <= sp + qbase + m) ? v : -1.0e30f;  // causal mask (+ pad keys)
      sc[m * KV_CAP + s] = v;
    }
  }
  __syncthreads();

  // --- Phase 2: per-row max / exp / sum (8 threads per row, float4 scans) -
  {
    int row = tid >> 3, sub = tid & 7;
    float* prow = sc + row * KV_CAP;
    int cnt = Spad >> 3;                   // elements per sub-thread (mult of 4)
    float4* p4 = (float4*)(prow + sub * cnt);
    int n4 = cnt >> 2;
    float mx = -1.0e30f;
    for (int j = 0; j < n4; ++j) {
      float4 v = p4[j];
      mx = fmaxf(mx, fmaxf(fmaxf(v.x, v.y), fmaxf(v.z, v.w)));
    }
    red8[row * 8 + sub] = mx;
    __syncthreads();
    if (sub == 0) {
      float m2 = red8[row * 8];
#pragma unroll
      for (int j = 1; j < 8; ++j) m2 = fmaxf(m2, red8[row * 8 + j]);
      red1[row] = m2;
    }
    __syncthreads();
    mx = red1[row];
    float sum = 0.0f;
    for (int j = 0; j < n4; ++j) {
      float4 v = p4[j];
      v.x = __expf(v.x - mx); v.y = __expf(v.y - mx);
      v.z = __expf(v.z - mx); v.w = __expf(v.w - mx);
      p4[j] = v;                           // store unnormalized probs
      sum += (v.x + v.y) + (v.z + v.w);
    }
    __syncthreads();                       // red8 reuse
    red8[row * 8 + sub] = sum;
    __syncthreads();
    if (sub == 0) {
      float t2 = 0.0f;
#pragma unroll
      for (int j = 0; j < 8; ++j) t2 += red8[row * 8 + j];
      red1[row] = 1.0f / t2;               // consumed at the final store
    }
    __syncthreads();
  }

  // --- Phase 3: ctx = P[16,Spad] * V[Spad,128] ----------------------------
  // Wave w handles output dim tiles {w, w+4}. V chunks staged in LDS.
  v8f co[2];
#pragma unroll
  for (int u = 0; u < 2; ++u)
#pragma unroll
    for (int i = 0; i < 8; ++i) co[u][i] = 0.0f;

  const int srow = tid >> 4;               // 0..7: staging row within octet
  const int scol = (tid & 15) * 8;         // 16B column chunk

  for (int k0 = 0; k0 < Spad; k0 += 32) {
    // Cooperative stage of V rows [k0, k0+32) x 128 dims (coalesced 16B).
#pragma unroll
    for (int rr = 0; rr < 4; ++rr) {
      int krow = k0 + rr * 8 + srow;
      const __bf16* g = vh +
          (((size_t)(b * KV_CAP + krow)) * NUM_KV_HEADS + kvh) * HEAD_DIM + scol;
      __bf16* l = vstage + (rr * 8 + srow) * HEAD_DIM + scol;
#if HAVE_ASYNC_LDS
      // LDS destination is passed as the low 32 bits of the generic pointer.
      __builtin_amdgcn_global_load_async_to_lds_b128(
          (gptr_v4i)(uintptr_t)g,
          (lptr_v4i)(uint32_t)(uintptr_t)l, 0, 0);
#else
      *(uint4*)l = *(const uint4*)g;
#endif
    }
#if HAVE_ASYNC_LDS
    wait_asynccnt0();
#endif
    __syncthreads();

    v16bf pa;                              // probs, A-matrix layout
#pragma unroll
    for (int i = 0; i < 8; ++i) {
      pa[i]     = (__bf16)sc[r * KV_CAP + k0 + hf * 16 + i];
      pa[i + 8] = (__bf16)sc[r * KV_CAP + k0 + hf * 16 + 8 + i];
    }
#pragma unroll
    for (int u = 0; u < 2; ++u) {
      int d0 = (wave + u * 4) * 16;
      v16bf vb;                            // V, B-matrix layout (from LDS)
#pragma unroll
      for (int i = 0; i < 16; ++i)
        vb[i] = vstage[(hf * 16 + i) * HEAD_DIM + d0 + r];
      co[u] = __builtin_amdgcn_wmma_f32_16x16x32_bf16(
          false, pa, false, vb, (short)0, co[u], false, false);
    }
    __syncthreads();                       // protect vstage for next chunk
  }

#pragma unroll
  for (int u = 0; u < 2; ++u) {
    int d0 = (wave + u * 4) * 16;
#pragma unroll
    for (int i = 0; i < 8; ++i) {
      int m = i + hf * 8;
      ctxh[((size_t)(b * SEQ + qbase + m)) * EMBED_DIM + h * HEAD_DIM + d0 + r] =
          (__bf16)(co[u][i] * red1[m]);    // deferred softmax normalization
    }
  }
}

// ---------------------------------------------------------------------------
// Host launcher
// ---------------------------------------------------------------------------
extern "C" void kernel_launch(void* const* d_in, const int* in_sizes, int n_in,
                              void* d_out, int out_size, void* d_ws, size_t ws_size,
                              hipStream_t stream) {
  (void)in_sizes; (void)n_in; (void)out_size; (void)ws_size;

  const float* x       = (const float*)d_in[0];
  const float* Wq      = (const float*)d_in[1];
  const float* Wk      = (const float*)d_in[2];
  const float* Wv      = (const float*)d_in[3];
  const float* Wo      = (const float*)d_in[4];
  const float* cache_k = (const float*)d_in[5];
  const float* cache_v = (const float*)d_in[6];
  const int*   spp     = (const int*)d_in[7];

  const size_t MT = (size_t)BATCH * SEQ;          // 2048 token rows
  char* ws = (char*)d_ws;
  size_t off = 0;
  auto alloc = [&](size_t bytes) {
    char* p = ws + off;
    off += (bytes + 255) & ~(size_t)255;
    return p;
  };
  __bf16* xh   = (__bf16*)alloc(MT * EMBED_DIM * 2);                    // 16 MB
  __bf16* Wqh  = (__bf16*)alloc((size_t)EMBED_DIM * EMBED_DIM * 2);     // 32 MB
  __bf16* Wkh  = (__bf16*)alloc((size_t)1024 * EMBED_DIM * 2);          //  8 MB
  __bf16* Wvh  = (__bf16*)alloc((size_t)1024 * EMBED_DIM * 2);          //  8 MB
  __bf16* Woh  = (__bf16*)alloc((size_t)EMBED_DIM * EMBED_DIM * 2);     // 32 MB
  float*  qf   = (float*) alloc(MT * EMBED_DIM * 4);                    // 32 MB
  float*  kf   = (float*) alloc(MT * 1024 * 4);                         //  8 MB
  float*  vf   = (float*) alloc(MT * 1024 * 4);                         //  8 MB
  __bf16* qh   = (__bf16*)alloc(MT * EMBED_DIM * 2);                    // 16 MB
  __bf16* kh   = (__bf16*)alloc((size_t)BATCH * KV_CAP * 1024 * 2);     //  8 MB
  __bf16* vh   = (__bf16*)alloc((size_t)BATCH * KV_CAP * 1024 * 2);     //  8 MB
  __bf16* ctxh = (__bf16*)qf;   // alias: qf is dead after rope_q        (16 MB)

  auto cvt = [&](const float* in, __bf16* out, int n) {
    cvt_f32_to_bf16<<<(n + 255) / 256, 256, 0, stream>>>(in, out, n);
  };

  // Stage 1: precision conversion (weights stay resident in 192 MB L2).
  cvt(x,       xh,  (int)(MT * EMBED_DIM));
  cvt(Wq,      Wqh, EMBED_DIM * EMBED_DIM);
  cvt(Wk,      Wkh, 1024 * EMBED_DIM);
  cvt(Wv,      Wvh, 1024 * EMBED_DIM);
  cvt(Wo,      Woh, EMBED_DIM * EMBED_DIM);
  cvt(cache_k, kh,  BATCH * KV_CAP * 1024);   // prior cache -> bf16 cache
  cvt(cache_v, vh,  BATCH * KV_CAP * 1024);

  // Stage 2: Q/K/V projections (WMMA GEMM, C = A * W^T).
  dim3 blk(256);
  gemm_nt_bf16<<<dim3(MT / 64, EMBED_DIM / 256), blk, 0, stream>>>(
      xh, Wqh, qf, (int)MT, EMBED_DIM, EMBED_DIM);
  gemm_nt_bf16<<<dim3(MT / 64, 1024 / 256), blk, 0, stream>>>(
      xh, Wkh, kf, (int)MT, 1024, EMBED_DIM);
  gemm_nt_bf16<<<dim3(MT / 64, 1024 / 256), blk, 0, stream>>>(
      xh, Wvh, vf, (int)MT, 1024, EMBED_DIM);

  // Stage 3: RoPE + cache writes (bf16).
  rope_q_kernel<<<(BATCH * SEQ * 32 * 64) / 256, 256, 0, stream>>>(qf, spp, qh);
  rope_k_kernel<<<(BATCH * SEQ * 8 * 64) / 256, 256, 0, stream>>>(kf, spp, kh);
  v_store_kernel<<<(BATCH * SEQ * 8 * 128) / 256, 256, 0, stream>>>(vf, spp, vh);

  // Stage 4: attention (dynamic LDS ~137 KB; CDNA5 WGP has 320 KB).
  const int smem = 16 * KV_CAP * 4 + 16 * 8 * 4 + 16 * 4 + 32 * HEAD_DIM * 2;
  (void)hipFuncSetAttribute((const void*)attn_kernel,
                            hipFuncAttributeMaxDynamicSharedMemorySize, smem);
  attn_kernel<<<BATCH * NUM_Q_HEADS * (SEQ / 16), 128, smem, stream>>>(
      qh, kh, vh, spp, ctxh);

  // Stage 5: output projection -> d_out (f32).
  gemm_nt_bf16<<<dim3(MT / 64, EMBED_DIM / 256), blk, 0, stream>>>(
      ctxh, Woh, (float*)d_out, (int)MT, EMBED_DIM, EMBED_DIM);
}
